// TransformerEncoderLayer_70729521430756
// MI455X (gfx1250) — matrix operand order repaired and checked by reading, not compiled, not run on hardware
//
#include <hip/hip_runtime.h>

// ---------------------------------------------------------------------------
// TransformerEncoderLayer (single-head SDPA) for MI455X / gfx1250, wave32.
//  - f32->bf16 convert (1/sqrt(E) folded into Wq)
//  - QKV projection GEMM via v_wmma_f32_16x16x32_bf16 (V stored transposed)
//  - flash attention: per-wave 16-query tile, Q cached in VGPRs, K/V tiles
//    staged in LDS per workgroup via the Tensor Data Mover (double-buffered,
//    s_wait_tensorcnt + barrier), online softmax, P relayout through LDS.
// ---------------------------------------------------------------------------

typedef __bf16 bf16;
typedef __attribute__((ext_vector_type(16))) __bf16 v16bf;
typedef __attribute__((ext_vector_type(8)))  __bf16 v8bf;
typedef __attribute__((ext_vector_type(4)))  __bf16 v4bf;
typedef __attribute__((ext_vector_type(8)))  float  v8f;
typedef __attribute__((ext_vector_type(4)))  unsigned int v4u;
typedef __attribute__((ext_vector_type(8)))  int    v8i;
typedef __attribute__((ext_vector_type(4)))  int    v4i;

#define BATCH 8
#define SEQ   2048
#define EMB   512

// LDS staging geometry (bytes). TDM pad_enable inserts 16B per row so
// fragment reads are spread across banks.
#define K_ROW_B   1040           // 512 bf16 + 16B pad   (banks: 4r mod 64)
#define K_BUF_B   (32 * K_ROW_B)
#define V_ROW_B   80             // 32 bf16 + 16B pad    (banks: 20r mod 64)
#define V_BUF_B   (512 * V_ROW_B)
#define SMEM_DYN  (2 * (K_BUF_B + V_BUF_B))   // 148480 B double-buffered

static __device__ __forceinline__ v8f zero8() {
  v8f z;
#pragma unroll
  for (int i = 0; i < 8; ++i) z[i] = 0.0f;
  return z;
}

static __device__ __forceinline__ v16bf cat8(v8bf lo, v8bf hi) {
  return __builtin_shufflevector(lo, hi, 0, 1, 2, 3, 4, 5, 6, 7,
                                          8, 9, 10, 11, 12, 13, 14, 15);
}

static __device__ __forceinline__ float redmax16(float v) {
  v = fmaxf(v, __shfl_xor(v, 1, 32));
  v = fmaxf(v, __shfl_xor(v, 2, 32));
  v = fmaxf(v, __shfl_xor(v, 4, 32));
  v = fmaxf(v, __shfl_xor(v, 8, 32));
  return v;
}
static __device__ __forceinline__ float redsum16(float v) {
  v += __shfl_xor(v, 1, 32);
  v += __shfl_xor(v, 2, 32);
  v += __shfl_xor(v, 4, 32);
  v += __shfl_xor(v, 8, 32);
  return v;
}

// ---------------------------------------------------------------------------
// Tensor Data Mover: 2-D tile load, descriptor per CDNA5 ISA cdna5_isa §8.
// g0: count=1 | lds_addr | global_addr(57b) | type=2
// g1: data_size=1(2B), pad_enable, pad_interval/amount, tensor dims/strides.
// ---------------------------------------------------------------------------
static __device__ __forceinline__ void tdm_load_2d(
    unsigned int lds_off, const void* gptr,
    unsigned int tensor_d0, unsigned int tensor_d1, unsigned long d0_stride,
    unsigned int tile_d0, unsigned int tile_d1,
    unsigned int pad_interval, unsigned int pad_amount) {
  unsigned long ga = (unsigned long)gptr;
  v4u g0;
  g0[0] = 1u;                                     // count=1, user mode
  g0[1] = lds_off;                                // lds_addr (bytes)
  g0[2] = (unsigned int)(ga & 0xffffffffu);
  g0[3] = (unsigned int)((ga >> 32) & 0x01ffffffu) | (2u << 30);  // type=2
  v8i g1;
  g1[0] = (int)((1u << 16) |                      // data_size = 2 bytes
                (1u << 20) |                      // pad_enable
                (pad_interval << 22) | (pad_amount << 25));
  g1[1] = (int)(tensor_d0 << 16);                 // [63:48] dim0 lo16
  g1[2] = (int)((tensor_d0 >> 16) | (tensor_d1 << 16));
  g1[3] = (int)((tensor_d1 >> 16) | (tile_d0 << 16));
  g1[4] = (int)(tile_d1 & 0xffffu);               // tile_dim1 (tile_dim2=0)
  g1[5] = (int)(unsigned int)(d0_stride & 0xffffffffu);
  g1[6] = (int)(unsigned int)((d0_stride >> 32) & 0xffffu);
  g1[7] = 0;
  v4i g2 = {0, 0, 0, 0};
  v4i g3 = {0, 0, 0, 0};
#if __clang_major__ >= 23
  __builtin_amdgcn_tensor_load_to_lds(g0, g1, g2, g3, g1, 0);  // 6-arg form
#else
  __builtin_amdgcn_tensor_load_to_lds(g0, g1, g2, g3, 0);      // ROCm 7.2
#endif
}

// ---------------------------------------------------------------------------
// f32 -> bf16 (optionally scaled), 4 elements/thread.
// ---------------------------------------------------------------------------
__global__ void cvt_f32_bf16(const float* __restrict__ x, bf16* __restrict__ o,
                             int n4, float scale) {
  int i = blockIdx.x * blockDim.x + threadIdx.x;
  if (i >= n4) return;
  const float4 f = reinterpret_cast<const float4*>(x)[i];
  v4bf b;
  b[0] = (bf16)(f.x * scale);
  b[1] = (bf16)(f.y * scale);
  b[2] = (bf16)(f.z * scale);
  b[3] = (bf16)(f.w * scale);
  reinterpret_cast<v4bf*>(o)[i] = b;
}

// ---------------------------------------------------------------------------
// Q/K/V projection: Y = X @ W^T.  blockIdx.z selects the weight.
// Wave tile: 32 rows x 64 cols (2x4 accumulators) -> B fragments reused 2x.
// which==2 (V) stores transposed per batch: Vt[(b*EMB + e)*SEQ + tok].
// ---------------------------------------------------------------------------
__global__ __launch_bounds__(256) void qkv_gemm(
    const bf16* __restrict__ X, const bf16* __restrict__ Wq,
    const bf16* __restrict__ Wk, const bf16* __restrict__ Wv,
    bf16* __restrict__ Q, bf16* __restrict__ K, bf16* __restrict__ Vt) {
  const int lane = threadIdx.x & 31;
  const int wave = threadIdx.x >> 5;
  const int l16  = lane & 15;
  const int hf   = lane >> 4;

  const int row0  = blockIdx.x * 256 + wave * 32;
  const int col0  = blockIdx.y * 64;
  const int which = blockIdx.z;
  const bf16* W = (which == 0) ? Wq : (which == 1) ? Wk : Wv;

  v8f acc[2][4];
#pragma unroll
  for (int m = 0; m < 2; ++m)
#pragma unroll
    for (int i = 0; i < 4; ++i) acc[m][i] = zero8();

  const bf16* Arow0 = X + (size_t)(row0 + l16) * EMB;
  const bf16* Arow1 = Arow0 + (size_t)16 * EMB;
#pragma unroll
  for (int es = 0; es < 16; ++es) {
    const int e = es * 32;
    v16bf a0 = cat8(*(const v8bf*)(Arow0 + e + hf * 8),
                    *(const v8bf*)(Arow0 + e + 16 + hf * 8));
    v16bf a1 = cat8(*(const v8bf*)(Arow1 + e + hf * 8),
                    *(const v8bf*)(Arow1 + e + 16 + hf * 8));
#pragma unroll
    for (int nf = 0; nf < 4; ++nf) {
      const int c = col0 + nf * 16 + l16;
      v16bf bm = cat8(*(const v8bf*)(W + (size_t)c * EMB + e + hf * 8),
                      *(const v8bf*)(W + (size_t)c * EMB + e + 16 + hf * 8));
      acc[0][nf] = __builtin_amdgcn_wmma_f32_16x16x32_bf16(
          false, a0, false, bm, (short)0, acc[0][nf], false, false);
      acc[1][nf] = __builtin_amdgcn_wmma_f32_16x16x32_bf16(
          false, a1, false, bm, (short)0, acc[1][nf], false, false);
    }
  }

#pragma unroll
  for (int m = 0; m < 2; ++m) {
    if (which == 2) {
#pragma unroll
      for (int nf = 0; nf < 4; ++nf) {
        const int c = col0 + nf * 16 + l16;
#pragma unroll
        for (int r = 0; r < 8; ++r) {
          const int row = row0 + m * 16 + r + 8 * hf;
          const int b   = row >> 11;
          const int t   = row & (SEQ - 1);
          Vt[((size_t)b * EMB + c) * SEQ + t] = (bf16)acc[m][nf][r];
        }
      }
    } else {
      bf16* O = (which == 0) ? Q : K;
#pragma unroll
      for (int nf = 0; nf < 4; ++nf) {
        const int c = col0 + nf * 16 + l16;
#pragma unroll
        for (int r = 0; r < 8; ++r) {
          const int row = row0 + m * 16 + r + 8 * hf;
          O[(size_t)row * EMB + c] = (bf16)acc[m][nf][r];
        }
      }
    }
  }
}

// ---------------------------------------------------------------------------
// Flash attention. One wave per 16-query tile; 8 waves/block share one batch
// and one key-block sequence, so K (32x512) and V^T (512x32) tiles are TDM-
// staged into LDS once per block and double-buffered.
// ---------------------------------------------------------------------------
__global__ __launch_bounds__(256) void flash_attn(
    const bf16* __restrict__ Q, const bf16* __restrict__ K,
    const bf16* __restrict__ Vt, const unsigned char* __restrict__ mask,
    float* __restrict__ out) {
  extern __shared__ __align__(16) char smem[];           // K/V tiles (dynamic)
  __shared__ __align__(16) bf16 plds[8][16][40];         // P relayout tiles

  const int lane = threadIdx.x & 31;
  const int wave = threadIdx.x >> 5;
  const int l16  = lane & 15;
  const int hf   = lane >> 4;

  const int gw = blockIdx.x * 8 + wave;     // all 8 waves: same batch
  const int b  = gw >> 7;
  const int q0 = (gw & 127) * 16;

  const bf16* Kbase = K  + (size_t)b * SEQ * EMB;
  const bf16* Vbase = Vt + (size_t)b * EMB * SEQ;
  const unsigned char* mrow = mask + b * SEQ;

  // LDS byte offset of the dynamic region (runtime addrspacecast is fine;
  // pointer ARRAYS of LDS-derived pointers are not - lld can't relocate them).
  const unsigned int smem_base = (unsigned int)(size_t)(void*)smem;

  // ---- cache Q fragments for the whole kernel (16 x v16bf = 128 VGPRs) ----
  const bf16* Qrow = Q + (size_t)(b * SEQ + q0 + l16) * EMB;
  v16bf qf[16];
#pragma unroll
  for (int es = 0; es < 16; ++es) {
    const int e = es * 32;
    qf[es] = cat8(*(const v8bf*)(Qrow + e + hf * 16),
                  *(const v8bf*)(Qrow + e + hf * 16 + 8));
  }

  v8f o[32];
#pragma unroll
  for (int i = 0; i < 32; ++i) o[i] = zero8();
  float mstat[8], lstat[8];
#pragma unroll
  for (int r = 0; r < 8; ++r) { mstat[r] = -1e30f; lstat[r] = 0.0f; }

  // ---- prologue: TDM-stage key block 0 ------------------------------------
  if (wave == 0) {
    // K tile: rows 0..31 x 512 (row-major); pad 4 DW per 256 DW row.
    tdm_load_2d(smem_base, Kbase,
                EMB, SEQ, EMB, EMB, 32, /*interval 2^(7+1)*8B=1024B*/ 7,
                /*amount 4 DW*/ 3);
    // V^T tile: 512 rows x keys 0..31; pad 4 DW per 16 DW row.
    tdm_load_2d(smem_base + 2 * K_BUF_B, Vbase,
                SEQ, EMB, SEQ, 32, EMB, /*interval 2^(3+1)*8B=64B*/ 3,
                /*amount 4 DW*/ 3);
  }

  for (int kb = 0; kb < SEQ; kb += 32) {
    const int cur = (kb >> 5) & 1;
    if (wave == 0) __builtin_amdgcn_s_wait_tensorcnt((short)0);
    __syncthreads();                        // tiles for `cur` visible to all
    if (wave == 0 && kb + 32 < SEQ) {       // prefetch next block via TDM
      const int nxt = cur ^ 1;
      tdm_load_2d(smem_base + nxt * K_BUF_B,
                  Kbase + (size_t)(kb + 32) * EMB,
                  EMB, SEQ, EMB, EMB, 32, 7, 3);
      tdm_load_2d(smem_base + 2 * K_BUF_B + nxt * V_BUF_B,
                  Vbase + kb + 32,
                  SEQ, EMB, SEQ, 32, EMB, 3, 3);
    }

    // ---- S = (Q*scale) @ K^T, 16x32 tile, K fragments from LDS ------------
    v8f s0 = zero8(), s1 = zero8();
    const char* k0 = smem + (size_t)cur * K_BUF_B + (size_t)l16 * K_ROW_B
                     + hf * 32;
    const char* k1 = k0 + 16 * K_ROW_B;
#pragma unroll
    for (int es = 0; es < 16; ++es) {
      const int eb = es * 64;               // 32 bf16 = 64 B per K step
      v16bf b0 = cat8(*(const v8bf*)(k0 + eb),
                      *(const v8bf*)(k0 + eb + 16));
      s0 = __builtin_amdgcn_wmma_f32_16x16x32_bf16(
          false, qf[es], false, b0, (short)0, s0, false, false);
      v16bf b1 = cat8(*(const v8bf*)(k1 + eb),
                      *(const v8bf*)(k1 + eb + 16));
      s1 = __builtin_amdgcn_wmma_f32_16x16x32_bf16(
          false, qf[es], false, b1, (short)0, s1, false, false);
    }

    // ---- key mask (column n <-> key kb + c*16 + l16) ----------------------
    const bool mk0 = mrow[kb + l16] != 0;
    const bool mk1 = mrow[kb + 16 + l16] != 0;
#pragma unroll
    for (int r = 0; r < 8; ++r) {
      if (mk0) s0[r] = -1e9f;
      if (mk1) s1[r] = -1e9f;
    }

    // ---- online softmax (fragment row r = query q0 + r + 8*hf) ------------
    float alpha[8];
#pragma unroll
    for (int r = 0; r < 8; ++r) {
      float rm = redmax16(fmaxf(s0[r], s1[r]));
      float mn = fmaxf(mstat[r], rm);
      alpha[r] = __expf(mstat[r] - mn);
      mstat[r] = mn;
      s0[r] = __expf(s0[r] - mn);
      s1[r] = __expf(s1[r] - mn);
      lstat[r] = lstat[r] * alpha[r] + redsum16(s0[r] + s1[r]);
    }
#pragma unroll
    for (int i = 0; i < 32; ++i) {
#pragma unroll
      for (int r = 0; r < 8; ++r) o[i][r] *= alpha[r];
    }

    // ---- relayout P (C-frag) -> A-frag via wave-private LDS tile ----------
#pragma unroll
    for (int r = 0; r < 8; ++r) {
      plds[wave][r + 8 * hf][l16]      = (bf16)s0[r];
      plds[wave][r + 8 * hf][16 + l16] = (bf16)s1[r];
    }
    v16bf pa = cat8(*(const v8bf*)&plds[wave][l16][hf * 16],
                    *(const v8bf*)&plds[wave][l16][hf * 16 + 8]);

    // ---- O += P @ V, V fragments from LDS ---------------------------------
    const char* vp = smem + 2 * K_BUF_B + (size_t)cur * V_BUF_B
                     + (size_t)l16 * V_ROW_B + hf * 32;
#pragma unroll
    for (int nf = 0; nf < 32; ++nf) {
      const char* vr = vp + (size_t)(nf * 16) * V_ROW_B;
      v16bf vb = cat8(*(const v8bf*)(vr), *(const v8bf*)(vr + 16));
      o[nf] = __builtin_amdgcn_wmma_f32_16x16x32_bf16(
          false, pa, false, vb, (short)0, o[nf], false, false);
    }
    __syncthreads();   // everyone done with `cur` before TDM reuses it
  }

  // ---- epilogue: divide by softmax denominator, store f32 -----------------
  float inv[8];
#pragma unroll
  for (int r = 0; r < 8; ++r) inv[r] = 1.0f / lstat[r];
  float* Orow = out + (size_t)(b * SEQ + q0) * EMB;
#pragma unroll
  for (int nf = 0; nf < 32; ++nf) {
#pragma unroll
    for (int r = 0; r < 8; ++r) {
      Orow[(size_t)(r + 8 * hf) * EMB + nf * 16 + l16] = o[nf][r] * inv[r];
    }
  }
}

// ---------------------------------------------------------------------------
// Launch. Workspace: X(16M) Q(16M) K(16M) V^T(16M) + bf16 weights (1.5M).
// ---------------------------------------------------------------------------
extern "C" void kernel_launch(void* const* d_in, const int* in_sizes, int n_in,
                              void* d_out, int out_size, void* d_ws,
                              size_t ws_size, hipStream_t stream) {
  const float* feat         = (const float*)d_in[0];
  const unsigned char* mask = (const unsigned char*)d_in[1];  // jnp.bool_: 1B
  const float* Wq           = (const float*)d_in[2];
  const float* Wk           = (const float*)d_in[3];
  const float* Wv           = (const float*)d_in[4];
  float* out                = (float*)d_out;

  char* ws = (char*)d_ws;
  bf16* Xb  = (bf16*)(ws);
  bf16* Qb  = (bf16*)(ws + ((size_t)16 << 20));
  bf16* Kb  = (bf16*)(ws + ((size_t)32 << 20));
  bf16* Vtb = (bf16*)(ws + ((size_t)48 << 20));
  bf16* Wqb = (bf16*)(ws + ((size_t)64 << 20));
  bf16* Wkb = Wqb + EMB * EMB;
  bf16* Wvb = Wkb + EMB * EMB;

  const int nX4 = BATCH * SEQ * EMB / 4;
  cvt_f32_bf16<<<(nX4 + 255) / 256, 256, 0, stream>>>(feat, Xb, nX4, 1.0f);

  const int nW4 = EMB * EMB / 4;
  const float scale = 0.044194173824159216f;  // 1/sqrt(512), folded into Wq
  cvt_f32_bf16<<<(nW4 + 255) / 256, 256, 0, stream>>>(Wq, Wqb, nW4, scale);
  cvt_f32_bf16<<<(nW4 + 255) / 256, 256, 0, stream>>>(Wk, Wkb, nW4, 1.0f);
  cvt_f32_bf16<<<(nW4 + 255) / 256, 256, 0, stream>>>(Wv, Wvb, nW4, 1.0f);

  dim3 gproj(BATCH * SEQ / 256, EMB / 64, 3);
  qkv_gemm<<<gproj, 256, 0, stream>>>(Xb, Wqb, Wkb, Wvb, Qb, Kb, Vtb);

  (void)hipFuncSetAttribute((const void*)flash_attn,
                            hipFuncAttributeMaxDynamicSharedMemorySize,
                            SMEM_DYN);
  flash_attn<<<(BATCH * SEQ / 16) / 8, 256, SMEM_DYN, stream>>>(
      Qb, Kb, Vtb, mask, out);
}
